// RpmEstimator_36661840839138
// MI455X (gfx1250) — matrix operand order repaired and checked by reading, not compiled, not run on hardware
//
#include <hip/hip_runtime.h>

typedef __attribute__((ext_vector_type(16))) _Float16 v16h;
typedef __attribute__((ext_vector_type(8)))  float    v8f;

#define N_FFT   8192
#define B_ROWS  4096
#define SPEC_W  2048          // bins stored per row (max needed harmonic bin = 1745)
#define BM      128
#define BN      128           // 64 freq bins * 2 (cos,sin interleaved columns)
#define BK      32
#define THREADS 256
#define NC      90            // number of RPM candidates (np.arange(27.6,29.1,1/60))
#define MH      60            // harmonics

union FragH { v16h h; uint4 u[2]; };

__global__ __launch_bounds__(256) void rpm_zero(float* __restrict__ rowsum) {
    int i = blockIdx.x * 256 + threadIdx.x;
    if (i < B_ROWS) rowsum[i] = 0.0f;
}

// One-time f32 -> f16 conversion of the signal (A16 is then L2-resident: 67 MB < 192 MB L2).
__global__ __launch_bounds__(256) void rpm_cvt(const float* __restrict__ sig,
                                               _Float16* __restrict__ a16) {
    size_t i = ((size_t)blockIdx.x * 256 + threadIdx.x) * 8;
    const float4* g = (const float4*)(sig + i);
    float4 f0 = g[0], f1 = g[1];
    union { _Float16 h[8]; uint4 u; } pk;
    pk.h[0] = (_Float16)f0.x; pk.h[1] = (_Float16)f0.y;
    pk.h[2] = (_Float16)f0.z; pk.h[3] = (_Float16)f0.w;
    pk.h[4] = (_Float16)f1.x; pk.h[5] = (_Float16)f1.y;
    pk.h[6] = (_Float16)f1.z; pk.h[7] = (_Float16)f1.w;
    *(uint4*)(a16 + i) = pk.u;
}

// DFT-as-GEMM: C[row, 2*bin+0/1] = sum_n sig[row,n] * {cos,sin}(2*pi*bin*n/8192)
// A tiles arrive via gfx1250 async global->LDS (ASYNCcnt); B tiles generated from a
// packed {cos,sin}-f16 LDS table. Fused epilogue: |X|, row-sum, store bins < 2048.
__global__ __launch_bounds__(THREADS) void rpm_dft_wmma(const _Float16* __restrict__ a16,
                                                        float* __restrict__ spec,
                                                        float* __restrict__ rowsum) {
    __shared__ unsigned  tableCS[N_FFT];    // {cos,sin}(2*pi*p/8192) packed f16x2, 32 KB
    __shared__ _Float16  Alds[BM * BK];     // [row][k], stride 32 halves, 8 KB
    __shared__ _Float16  Blds[BN * BK];     // [col][k], stride 32 halves, 8 KB

    const int tid   = threadIdx.x;
    const int lane  = tid & 31;
    const int wid   = tid >> 5;             // 0..7
    const int waveM = wid >> 1;             // 0..3 -> 32-row stripes
    const int waveN = wid & 1;              // 0..1 -> 64-col stripes

    const int nBase = blockIdx.x * BN;      // global column base (2*bin + re/im)
    const int mBase = blockIdx.y * BM;      // global row base

    // Build packed cos/sin table once.
    for (int i = tid; i < N_FFT; i += THREADS) {
        float ang = (float)i * 7.6699039394282222e-4f;   // 2*pi/8192
        union { _Float16 h[2]; unsigned u; } p;
        p.h[0] = (_Float16)cosf(ang);
        p.h[1] = (_Float16)sinf(ang);
        tableCS[i] = p.u;
    }
    __syncthreads();

    // A-tile async loader mapping: 2 threads per row, 16 consecutive f16 (32 B) each.
    const int aRow = tid >> 1;              // 0..127
    const int aKh  = (tid & 1) * 16;        // 0 or 16 (halves)
    const _Float16* aPtr = a16 + (size_t)(mBase + aRow) * N_FFT + aKh;
    const unsigned  aLds = (unsigned)(uintptr_t)&Alds[aRow * BK + aKh];

    // B-tile generator mapping: one bin (cos+sin column pair) x 8 k values.
    const int bBinL = tid & 63;             // local bin 0..63
    const int bK8   = (tid >> 6) * 8;       // 0,8,16,24
    const int bBin  = (nBase >> 1) + bBinL; // global bin

    const v8f vzero = {0.f, 0.f, 0.f, 0.f, 0.f, 0.f, 0.f, 0.f};
    v8f acc[2][4];
#pragma unroll
    for (int a = 0; a < 2; ++a)
#pragma unroll
        for (int b = 0; b < 4; ++b) acc[a][b] = vzero;

    const int half = lane & 15;
    const int hiA  = (lane < 16) ? 0 : 8;   // ISA A 16x32 f16 layout (7.12.2)
    const int hiB  = (lane < 16) ? 0 : 8;   // ISA B 32x16 f16 layout (same K order as A)

    for (int kb = 0; kb < N_FFT; kb += BK) {
        // ---- A tile: async global->LDS B128 x2 (offset applies to both sides) ----
        {
            unsigned long long ga = (unsigned long long)(uintptr_t)(aPtr + kb);
            asm volatile(
                "global_load_async_to_lds_b128 %0, %1, off\n\t"
                "global_load_async_to_lds_b128 %0, %1, off offset:16"
                :: "v"(aLds), "v"(ga) : "memory");
        }
        // ---- B tile: 8 packed {cos,sin} gathers -> 16 f16 (one Re col + one Im col) ----
        {
            int phase = (bBin * (kb + bK8)) & (N_FFT - 1);
            union { _Float16 h[8]; uint4 u; } cpk, spk;
#pragma unroll
            for (int j = 0; j < 8; ++j) {
                union { _Float16 h[2]; unsigned u; } p;
                p.u = tableCS[phase];
                cpk.h[j] = p.h[0];
                spk.h[j] = p.h[1];
                phase = (phase + bBin) & (N_FFT - 1);
            }
            *(uint4*)&Blds[(2 * bBinL + 0) * BK + bK8] = cpk.u;
            *(uint4*)&Blds[(2 * bBinL + 1) * BK + bK8] = spk.u;
        }
        asm volatile("s_wait_asynccnt 0x0" ::: "memory");
        __syncthreads();

        // ---- fragments + 8 WMMAs per wave ----
        FragH afrag[2];
#pragma unroll
        for (int wm = 0; wm < 2; ++wm) {
            const _Float16* src = &Alds[(waveM * 32 + wm * 16 + half) * BK + hiA];
            afrag[wm].u[0] = *(const uint4*)(src);        // K {hiA..hiA+7}
            afrag[wm].u[1] = *(const uint4*)(src + 16);   // K {hiA+16..hiA+23}
        }
#pragma unroll
        for (int cn = 0; cn < 4; ++cn) {
            FragH bfrag;
            const _Float16* src = &Blds[(waveN * 64 + cn * 16 + half) * BK + hiB];
            bfrag.u[0] = *(const uint4*)(src);            // K {hiB..hiB+7}
            bfrag.u[1] = *(const uint4*)(src + 16);       // K {hiB+16..hiB+23}
#pragma unroll
            for (int wm = 0; wm < 2; ++wm)
                acc[wm][cn] = __builtin_amdgcn_wmma_f32_16x16x32_f16(
                    false, afrag[wm].h, false, bfrag.h,
                    (short)0, acc[wm][cn], false, false);
        }
        __syncthreads();
    }

    // ---- fused epilogue: |X|, spectrum store, row-sum for threshold ----
    const int rowOff = (lane & 16) ? 8 : 0;   // C layout: VGPR r -> M = r (+8 upper half)
#pragma unroll
    for (int wm = 0; wm < 2; ++wm) {
        float rp[8];
#pragma unroll
        for (int r = 0; r < 8; ++r) rp[r] = 0.0f;
#pragma unroll
        for (int cn = 0; cn < 4; ++cn) {
            const int colg = nBase + waveN * 64 + cn * 16 + half;
            const int bin2 = colg >> 1;
#pragma unroll
            for (int r = 0; r < 8; ++r) {
                float c  = acc[wm][cn][r];
                float sq = c * c;
                float mag = sqrtf(sq + __shfl_xor(sq, 1, 32));  // re/im in adjacent lanes
                rp[r] += mag;                                    // each bin counted twice
                int rowg = mBase + waveM * 32 + wm * 16 + r + rowOff;
                if (((lane & 1) == 0) && bin2 < SPEC_W)
                    spec[(size_t)rowg * SPEC_W + bin2] = mag;
            }
        }
#pragma unroll
        for (int r = 0; r < 8; ++r) {
            float s = rp[r];
            s += __shfl_xor(s, 1, 32);
            s += __shfl_xor(s, 2, 32);
            s += __shfl_xor(s, 4, 32);
            s += __shfl_xor(s, 8, 32);     // 16-lane half reduction
            if ((lane & 15) == 0) {
                int rowg = mBase + waveM * 32 + wm * 16 + r + rowOff;
                atomicAdd(&rowsum[rowg], s * 0.5f);   // /2: each bin counted twice
            }
        }
    }
}

// One block per batch row: threshold + harmonic fail count + argmin-first.
__global__ __launch_bounds__(128) void rpm_decide(const float* __restrict__ spec,
                                                  const float* __restrict__ rowsum,
                                                  float* __restrict__ out) {
    __shared__ int keys[128];
    const int b = blockIdx.x;
    const int c = threadIdx.x;
    const float* srow = spec + (size_t)b * SPEC_W;
    const float thr = rowsum[b] * (1.5f / 4096.0f);   // mean over half spectrum * 1.5

    int key = 0x7FFFFFFF;
    if (c < NC) {
        float cand = (float)(27.6 + (double)c * (1.0 / 60.0));  // np.arange->f32 semantics
        int fails = 0;
        for (int k = 1; k <= MH; ++k) {
            int id = (int)((float)k * cand);          // f32 mul, trunc -> bin index
            fails += (srow[id] > thr) ? 0 : 1;
        }
        key = (fails << 8) | c;   // argmax(2^-fails) == first min-fails index
    }
    keys[c] = key;
    __syncthreads();
    for (int s = 64; s > 0; s >>= 1) {
        if (c < s) keys[c] = min(keys[c], keys[c + s]);
        __syncthreads();
    }
    if (c == 0) {
        int best = keys[0] & 255;
        float cand = (float)(27.6 + (double)best * (1.0 / 60.0));
        out[b] = cand * 60.0f;
    }
}

extern "C" void kernel_launch(void* const* d_in, const int* in_sizes, int n_in,
                              void* d_out, int out_size, void* d_ws, size_t ws_size,
                              hipStream_t stream) {
    (void)in_sizes; (void)n_in; (void)out_size; (void)ws_size;
    const float* sig = (const float*)d_in[0];
    // ws layout: spec [4096 x 2048] f32 (32 MB) | rowsum [4096] f32 | A16 [4096 x 8192] f16 (64 MB)
    float*    spec   = (float*)d_ws;
    float*    rowsum = (float*)((char*)d_ws + (size_t)B_ROWS * SPEC_W * sizeof(float));
    _Float16* a16    = (_Float16*)((char*)d_ws + (size_t)B_ROWS * SPEC_W * sizeof(float)
                                              + (size_t)B_ROWS * sizeof(float));
    float* out = (float*)d_out;

    rpm_zero<<<16, 256, 0, stream>>>(rowsum);
    rpm_cvt<<<(B_ROWS * (N_FFT / 8)) / 256, 256, 0, stream>>>(sig, a16);
    dim3 grid(N_FFT / BN, B_ROWS / BM);   // 64 x 32 output tiles
    rpm_dft_wmma<<<grid, THREADS, 0, stream>>>(a16, spec, rowsum);
    rpm_decide<<<B_ROWS, 128, 0, stream>>>(spec, rowsum, out);
}
